// EfficientTransformerBlock_6176162972199
// MI455X (gfx1250) — matrix operand order repaired and checked by reading, not compiled, not run on hardware
//
#include <hip/hip_runtime.h>
#include <hip/hip_bf16.h>
#include <math.h>

typedef __bf16 bf16_t;
typedef __attribute__((ext_vector_type(16))) __bf16 v16bf;
typedef __attribute__((ext_vector_type(8)))  __bf16 v8bf;
typedef __attribute__((ext_vector_type(8)))  float  v8f;

#define USE_ASYNC_LDS 1   // gfx1250 async global->LDS staging

// ---- problem dims (compile-time; H=W=T=48 per setup_inputs) ----
constexpr int   CC     = 96;
constexpr int   C3     = 288;
constexpr int   NH     = 3;
constexpr int   DH     = 1536;     // B_*hd = 48*32
constexpr int   NTOK   = 2304;     // H*W
constexpr int   TT     = 48;       // T (and B_)
constexpr int   LL     = 110592;   // H*W*T
constexpr int   MLPH   = 384;
constexpr float QSCALE = 0.17677669529663689f;  // 32^-0.5
constexpr float LNEPS  = 1e-5f;

// ---- workspace layout (bytes), lifetime-aliased (~183 MiB total) ----
constexpr size_t OFF_HPERM = 0;                                    // L*C bf16 ; reused as OBUF
constexpr size_t OFF_QKVB  = OFF_HPERM + (size_t)LL * CC * 2;      // L*288 bf16 ; reused as S(f32), then M1
constexpr size_t OFF_Q     = OFF_QKVB  + (size_t)LL * C3 * 2;      // NH*NTOK*DH bf16 ; P lives here later
constexpr size_t OFF_K     = OFF_Q     + (size_t)NH * NTOK * DH * 2;   // K^T [h][d][n]
constexpr size_t OFF_V     = OFF_K     + (size_t)NH * NTOK * DH * 2;   // reused as H2
constexpr size_t OFF_X2    = OFF_V     + (size_t)NH * NTOK * DH * 2;   // L*C f32
constexpr size_t OFF_WQKV  = OFF_X2    + (size_t)LL * CC * 4;          // transposed bf16 weights
constexpr size_t OFF_WPROJ = OFF_WQKV  + (size_t)C3 * CC * 2;
constexpr size_t OFF_WFC1  = OFF_WPROJ + (size_t)CC * CC * 2;
constexpr size_t OFF_WFC2  = OFF_WFC1  + (size_t)MLPH * CC * 2;

// ---- helpers ----
__device__ __forceinline__ v16bf ld_frag16(const bf16_t* p0, const bf16_t* p1) {
    union { v16bf v; v8bf h[2]; } u;
    u.h[0] = *(const v8bf*)p0;
    u.h[1] = *(const v8bf*)p1;
    return u.v;
}
__device__ __forceinline__ float wred_sum(float v) {
    #pragma unroll
    for (int o = 16; o > 0; o >>= 1) v += __shfl_xor(v, o, 32);
    return v;
}
__device__ __forceinline__ float wred_max(float v) {
    #pragma unroll
    for (int o = 16; o > 0; o >>= 1) v = fmaxf(v, __shfl_xor(v, o, 32));
    return v;
}

// 16B global -> LDS copy on the gfx1250 ASYNCcnt path
__device__ __forceinline__ void cp16(const bf16_t* g, bf16_t* l) {
#if USE_ASYNC_LDS
    unsigned lo = (unsigned)(size_t)l;   // generic LDS addr -> wave LDS byte offset (addr[31:0])
    asm volatile("global_load_async_to_lds_b128 %0, %1, off"
                 :: "v"(lo), "v"(g) : "memory");
#else
    *(v8bf*)l = *(const v8bf*)g;
#endif
}
__device__ __forceinline__ void cp_wait() {
#if USE_ASYNC_LDS
#if defined(__has_builtin)
#if __has_builtin(__builtin_amdgcn_s_wait_asynccnt)
    __builtin_amdgcn_s_wait_asynccnt(0);
#else
    asm volatile("s_wait_asynccnt 0x0" ::: "memory");
#endif
#else
    asm volatile("s_wait_asynccnt 0x0" ::: "memory");
#endif
#endif
}

// ---- weight fp32 -> bf16 + transpose to [K][N] so every GEMM is NN ----
__global__ __launch_bounds__(256) void wcvt_k(const float* __restrict__ qkvw,
                                              const float* __restrict__ projw,
                                              const float* __restrict__ fc1w,
                                              const float* __restrict__ fc2w,
                                              bf16_t* __restrict__ oq, bf16_t* __restrict__ op,
                                              bf16_t* __restrict__ o1, bf16_t* __restrict__ o2) {
    int i = blockIdx.x * 256 + threadIdx.x;      // 0 .. 110591
    if (i < 27648) {            // qkv_w [288][96] -> [96][288]
        int n = i / 96, k = i % 96;  oq[k * C3 + n] = (bf16_t)qkvw[i];
    } else if (i < 36864) {     // proj_w [96][96] -> [96][96]
        int j = i - 27648; int n = j / 96, k = j % 96;  op[k * CC + n] = (bf16_t)projw[j];
    } else if (i < 73728) {     // fc1_w [384][96] -> [96][384]
        int j = i - 36864; int n = j / 96, k = j % 96;  o1[k * MLPH + n] = (bf16_t)fc1w[j];
    } else {                    // fc2_w [96][384] -> [384][96]
        int j = i - 73728; int n = j / 384, k = j % 384; o2[k * CC + n] = (bf16_t)fc2w[j];
    }
}

// ---- LayerNorm: one wave32 per row of 96, 3 elems/lane; PERM writes axial layout ----
template<int PERM>
__global__ __launch_bounds__(256) void ln_kernel(const float* __restrict__ x,
                                                 const float* __restrict__ g,
                                                 const float* __restrict__ b,
                                                 bf16_t* __restrict__ out) {
    const int wv = threadIdx.x >> 5, lane = threadIdx.x & 31;
    const int row = blockIdx.x * 8 + wv;
    const float* xr = x + (size_t)row * CC;
    const int c0 = lane * 3;
    float a0 = xr[c0], a1 = xr[c0 + 1], a2 = xr[c0 + 2];
    float mean = wred_sum(a0 + a1 + a2) * (1.0f / CC);
    float d0 = a0 - mean, d1 = a1 - mean, d2 = a2 - mean;
    float var = wred_sum(d0 * d0 + d1 * d1 + d2 * d2) * (1.0f / CC);
    float rstd = rsqrtf(var + LNEPS);
    size_t drow;
    if (PERM) { int hw = row / TT, t = row % TT; drow = (size_t)t * NTOK + hw; }
    else      { drow = row; }
    bf16_t* orow = out + drow * CC;
    orow[c0]     = (bf16_t)(d0 * rstd * g[c0]     + b[c0]);
    orow[c0 + 1] = (bf16_t)(d1 * rstd * g[c0 + 1] + b[c0 + 1]);
    orow[c0 + 2] = (bf16_t)(d2 * rstd * g[c0 + 2] + b[c0 + 2]);
}

// ---- unified bf16 NN GEMM: out[m,n] = sum_k A[m,k]*B[k,n] (+bias)(+GELU)(+resid)(scatter) ----
// block = 8 waves; block tile 256(M) x 96(N); wave tile 32x96; K-step 32 -> 12 WMMA/step/wave.
// Async double-buffered LDS staging, one barrier per K-step, all frags hoisted before WMMA chain.
template<int OUT_BF16, int ACT_GELU, int HAS_BIAS, int HAS_RES, int SCATTER_O>
__global__ __launch_bounds__(256) void gemm_nn(const bf16_t* __restrict__ A,
                                               const bf16_t* __restrict__ B,
                                               const float* __restrict__ bias,
                                               const float* __restrict__ resid,
                                               void* __restrict__ outv,
                                               int M, int N, int K,
                                               size_t strideA, size_t strideB, size_t strideO) {
    __shared__ bf16_t As[2][256 * 32];
    __shared__ bf16_t Bs[2][32 * 96];
    const int t = threadIdx.x, wv = t >> 5, lane = t & 31;
    const int nBase = blockIdx.x * 96;
    const int mBase = blockIdx.y * 256;
    const int bat   = blockIdx.z;
    A += (size_t)bat * strideA;
    B += (size_t)bat * strideB;

    v8f z = {};
    v8f acc[2][6];
    #pragma unroll
    for (int i = 0; i < 2; i++)
        #pragma unroll
        for (int j = 0; j < 6; j++) acc[i][j] = z;

    // per-thread staging offsets (K-loop invariant)
    const bf16_t* gA = A + (size_t)(mBase + t) * K;       // + k0 per step
    bf16_t* lA0 = &As[0][t * 32];
    bf16_t* lA1 = &As[1][t * 32];
    const int c0i = t, c1i = t + 192;                      // B chunks for t<192
    const int bk0 = c0i / 12, bn0 = (c0i % 12) * 8;
    const int bk1 = c1i / 12, bn1 = (c1i % 12) * 8;
    const bf16_t* gB0 = B + (size_t)bk0 * N + nBase + bn0; // + k0*N per step
    const bf16_t* gB1 = B + (size_t)bk1 * N + nBase + bn1;
    bf16_t* lB0[2] = { &Bs[0][bk0 * 96 + bn0], &Bs[1][bk0 * 96 + bn0] };
    bf16_t* lB1[2] = { &Bs[0][bk1 * 96 + bn1], &Bs[1][bk1 * 96 + bn1] };

    auto stage = [&](int buf, int k0) {
        const bf16_t* ga = gA + k0;
        bf16_t* la = buf ? lA1 : lA0;
        #pragma unroll
        for (int c = 0; c < 4; c++) cp16(ga + c * 8, la + c * 8);
        if (t < 192) {
            size_t koff = (size_t)k0 * N;
            cp16(gB0 + koff, lB0[buf]);
            cp16(gB1 + koff, lB1[buf]);
        }
    };

    stage(0, 0);
    cp_wait();
    __syncthreads();

    const int nk = K / 32;
    const int m0 = wv * 32;
    const int mr = lane & 15, hs = (lane >> 4) * 8;
    for (int ik = 0; ik < nk; ik++) {
        const int buf = ik & 1;
        if (ik + 1 < nk) stage(buf ^ 1, (ik + 1) * 32);
        const bf16_t* as = &As[buf][0];
        const bf16_t* bs = &Bs[buf][0];
        // hoist ALL fragments -> one DS drain, then 12 independent back-to-back WMMAs
        v16bf a0 = ld_frag16(as + (m0 + mr) * 32 + hs,      as + (m0 + mr) * 32 + 16 + hs);
        v16bf a1 = ld_frag16(as + (m0 + 16 + mr) * 32 + hs, as + (m0 + 16 + mr) * 32 + 16 + hs);
        v16bf bfr[6];
        #pragma unroll
        for (int j = 0; j < 6; j++) {
            #pragma unroll
            for (int s = 0; s < 16; s++) {
                const int kk = (s < 8) ? (hs + s) : (16 + hs + (s - 8));
                bfr[j][s] = bs[kk * 96 + j * 16 + mr];
            }
        }
        #pragma unroll
        for (int j = 0; j < 6; j++) {
            acc[0][j] = __builtin_amdgcn_wmma_f32_16x16x32_bf16(false, a0, false, bfr[j], (short)0, acc[0][j], false, false);
            acc[1][j] = __builtin_amdgcn_wmma_f32_16x16x32_bf16(false, a1, false, bfr[j], (short)0, acc[1][j], false, false);
        }
        cp_wait();        // next tile's async copies complete
        __syncthreads();  // all waves' copies visible; reads of 'buf' done before overwrite
    }

    // C/D layout: VGPR r -> M = r (lanes 0-15) / r+8 (lanes 16-31); N = lane&15
    const int nc = lane & 15, mo = (lane >> 4) << 3;
    #pragma unroll
    for (int i = 0; i < 2; i++) {
        #pragma unroll
        for (int j = 0; j < 6; j++) {
            #pragma unroll
            for (int r = 0; r < 8; r++) {
                int mm = mBase + wv * 32 + i * 16 + mo + r;
                int nn = nBase + j * 16 + nc;
                float v = acc[i][j][r];
                if (HAS_BIAS) v += bias[nn];
                if (ACT_GELU) v = 0.5f * v * (1.0f + erff(v * 0.70710678118f));
                if (SCATTER_O) {
                    // o: [1,nh,N,1536] -> transpose(0,2,1,3) -> reshape(48,2304,96) flat [L,96]
                    int g = bat * DH + nn;
                    size_t l = (size_t)(mm / TT) * NTOK + (size_t)(mm % TT) * TT + (size_t)(g / CC);
                    ((bf16_t*)outv)[l * CC + (g % CC)] = (bf16_t)v;
                } else {
                    size_t o = (size_t)mm * N + nn;
                    if (HAS_RES) v += resid[o];
                    if (OUT_BF16) ((bf16_t*)outv + (size_t)bat * strideO)[o] = (bf16_t)v;
                    else          ((float*)outv  + (size_t)bat * strideO)[o] = v;
                }
            }
        }
    }
}

// ---- split qkv buffer [48][2304][288] into Q(scaled)/K^T/V ----
__global__ __launch_bounds__(256) void repack_k(const bf16_t* __restrict__ qkvb,
                                                bf16_t* __restrict__ Q,
                                                bf16_t* __restrict__ KT,
                                                bf16_t* __restrict__ Vb) {
    size_t idx = (size_t)blockIdx.x * 256 + threadIdx.x;   // < NH*NTOK*DH
    int h0 = (int)(idx / ((size_t)NTOK * DH));
    size_t rem = idx % ((size_t)NTOK * DH);
    int n = (int)(rem / DH), d = (int)(rem % DH);
    int g = h0 * DH + d;
    int b16 = g / C3, c3 = g % C3;
    size_t blk = (size_t)NTOK * C3;
    size_t rq = ((size_t)b16 * NTOK + n) * C3 + c3;
    Q[idx]  = (bf16_t)((float)qkvb[rq] * QSCALE);                       // [h][n][d]
    KT[((size_t)h0 * DH + d) * NTOK + n] = qkvb[rq + 16 * blk];         // [h][d][n]
    Vb[idx] = qkvb[rq + 32 * blk];                                      // [h][n][d]
}

// ---- softmax over rows of S [NH*NTOK][NTOK], writes P bf16 ----
__global__ __launch_bounds__(256) void softmax_k(const float* __restrict__ S,
                                                 bf16_t* __restrict__ P) {
    __shared__ float red[8];
    const int t = threadIdx.x, wv = t >> 5, lane = t & 31;
    const float* row = S + (size_t)blockIdx.x * NTOK;
    bf16_t* prow = P + (size_t)blockIdx.x * NTOK;
    float v[9];
    float mx = -3.0e38f;
    #pragma unroll
    for (int i = 0; i < 9; i++) { v[i] = row[t + i * 256]; mx = fmaxf(mx, v[i]); }
    mx = wred_max(mx);
    if (lane == 0) red[wv] = mx;
    __syncthreads();
    float bm = red[0];
    #pragma unroll
    for (int j = 1; j < 8; j++) bm = fmaxf(bm, red[j]);
    __syncthreads();
    float sum = 0.f;
    #pragma unroll
    for (int i = 0; i < 9; i++) { v[i] = __expf(v[i] - bm); sum += v[i]; }
    sum = wred_sum(sum);
    if (lane == 0) red[wv] = sum;
    __syncthreads();
    float tot = 0.f;
    #pragma unroll
    for (int j = 0; j < 8; j++) tot += red[j];
    float inv = 1.0f / tot;
    #pragma unroll
    for (int i = 0; i < 9; i++) prow[t + i * 256] = (bf16_t)(v[i] * inv);
}

extern "C" void kernel_launch(void* const* d_in, const int* in_sizes, int n_in,
                              void* d_out, int out_size, void* d_ws, size_t ws_size,
                              hipStream_t stream) {
    (void)in_sizes; (void)n_in; (void)out_size; (void)ws_size;
    const float* x      = (const float*)d_in[0];
    const float* ln1_g  = (const float*)d_in[2];
    const float* ln1_b  = (const float*)d_in[3];
    const float* qkv_w  = (const float*)d_in[4];
    const float* qkv_b  = (const float*)d_in[5];
    const float* proj_w = (const float*)d_in[6];
    const float* proj_b = (const float*)d_in[7];
    const float* ln2_g  = (const float*)d_in[8];
    const float* ln2_b  = (const float*)d_in[9];
    const float* fc1_w  = (const float*)d_in[10];
    const float* fc1_b  = (const float*)d_in[11];
    const float* fc2_w  = (const float*)d_in[12];
    const float* fc2_b  = (const float*)d_in[13];

    char* ws = (char*)d_ws;
    bf16_t* hperm = (bf16_t*)(ws + OFF_HPERM);
    bf16_t* qkvb  = (bf16_t*)(ws + OFF_QKVB);
    bf16_t* Qb    = (bf16_t*)(ws + OFF_Q);
    bf16_t* KT    = (bf16_t*)(ws + OFF_K);
    bf16_t* Vb    = (bf16_t*)(ws + OFF_V);
    float*  Sb    = (float*)(ws + OFF_QKVB);   // alias over dead qkvb
    bf16_t* Pb    = (bf16_t*)(ws + OFF_Q);     // alias over dead Q/K
    bf16_t* obuf  = (bf16_t*)(ws + OFF_HPERM); // alias over dead hperm
    float*  x2    = (float*)(ws + OFF_X2);
    bf16_t* h2    = (bf16_t*)(ws + OFF_V);     // alias over dead V
    bf16_t* m1    = (bf16_t*)(ws + OFF_QKVB);  // alias over dead S + P head
    bf16_t* wqkvT = (bf16_t*)(ws + OFF_WQKV);
    bf16_t* wprojT= (bf16_t*)(ws + OFF_WPROJ);
    bf16_t* wfc1T = (bf16_t*)(ws + OFF_WFC1);
    bf16_t* wfc2T = (bf16_t*)(ws + OFF_WFC2);
    float*  out   = (float*)d_out;

    // 1) weights -> bf16, transposed to [K][N]
    wcvt_k<<<432, 256, 0, stream>>>(qkv_w, proj_w, fc1_w, fc2_w, wqkvT, wprojT, wfc1T, wfc2T);
    // 2) LN1 + axial permute to [T, H*W, C] bf16
    ln_kernel<1><<<LL / 8, 256, 0, stream>>>(x, ln1_g, ln1_b, hperm);
    // 3) QKV GEMM: [110592 x 96] @ [96 x 288] (+bias) -> bf16
    gemm_nn<1, 0, 1, 0, 0><<<dim3(C3 / 96, LL / 256, 1), 256, 0, stream>>>(
        hperm, wqkvT, qkv_b, nullptr, qkvb, LL, C3, CC, 0, 0, 0);
    // 4) split into Q(scaled)[h][n][d] / K^T[h][d][n] / V[h][n][d]
    repack_k<<<(NH * NTOK * DH) / 256, 256, 0, stream>>>(qkvb, Qb, KT, Vb);
    // 5) S = Q @ K^T per head: [2304 x 1536] @ [1536 x 2304] -> f32
    gemm_nn<0, 0, 0, 0, 0><<<dim3(NTOK / 96, NTOK / 256, NH), 256, 0, stream>>>(
        Qb, KT, nullptr, nullptr, Sb, NTOK, NTOK, DH,
        (size_t)NTOK * DH, (size_t)DH * NTOK, (size_t)NTOK * NTOK);
    // 6) row softmax -> P bf16
    softmax_k<<<NH * NTOK, 256, 0, stream>>>(Sb, Pb);
    // 7) O = P @ V with fused torch-faithful output reshape scatter
    gemm_nn<1, 0, 0, 0, 1><<<dim3(DH / 96, NTOK / 256, NH), 256, 0, stream>>>(
        Pb, Vb, nullptr, nullptr, obuf, NTOK, DH, NTOK,
        (size_t)NTOK * NTOK, (size_t)NTOK * DH, 0);
    // 8) proj (+bias) + residual with shortcut x -> x2 f32
    gemm_nn<0, 0, 1, 1, 0><<<dim3(CC / 96, LL / 256, 1), 256, 0, stream>>>(
        obuf, wprojT, proj_b, x, x2, LL, CC, CC, 0, 0, 0);
    // 9) LN2 -> h2 bf16
    ln_kernel<0><<<LL / 8, 256, 0, stream>>>(x2, ln2_g, ln2_b, h2);
    // 10) fc1 (+bias, exact GELU) -> m1 bf16
    gemm_nn<1, 1, 1, 0, 0><<<dim3(MLPH / 96, LL / 256, 1), 256, 0, stream>>>(
        h2, wfc1T, fc1_b, nullptr, m1, LL, MLPH, CC, 0, 0, 0);
    // 11) fc2 (+bias) + residual x2 -> d_out f32
    gemm_nn<0, 0, 1, 1, 0><<<dim3(CC / 96, LL / 256, 1), 256, 0, stream>>>(
        m1, wfc2T, fc2_b, x2, out, LL, CC, MLPH, 0, 0, 0);
}
